// INSGT_SL_55396488184200
// MI455X (gfx1250) — matrix operand order repaired and checked by reading, not compiled, not run on hardware
//
#include <hip/hip_runtime.h>
#include <hip/hip_fp16.h>

typedef __attribute__((ext_vector_type(16))) _Float16 v16h;
typedef __attribute__((ext_vector_type(8)))  float    v8f;

#define F1C   128
#define MC    256
#define SC    32
#define BC    8
#define NC    8192
#define LMAXC 1024
#define LENC  131072
#define ROWS  (SC * BC * F1C)               // 32768 independent 256-pt DFT rows
#define TEMP_BYTES ((size_t)ROWS * MC * 8)  // float2 spectrum per row
#define TWID_ELEMS (8 * 16 * 32 * 16)       // 8 K-chunks x 16 col-tiles x 32 lanes x 16 halves

// Low 32 bits of a generic pointer to __shared__ = LDS byte offset (ISA 10.2).
__device__ __forceinline__ unsigned lds_offset(const void* p) {
    return (unsigned)(unsigned long long)p;
}

// ---------------------------------------------------------------------------
// Kernel 0: zero the accumulated output and build the DFT twiddle tables,
// pre-swizzled into this code's contiguous-K B-fragment order (A uses the same):
//   lane L holds N = L%16, K = 16*(L/16) + e  (e = 0..15) within a 32-K chunk.
// ---------------------------------------------------------------------------
__global__ void insgt_init(float* __restrict__ out,
                           _Float16* __restrict__ cosT,
                           _Float16* __restrict__ sinT) {
    int tid = blockIdx.x * blockDim.x + threadIdx.x;
    if (tid < BC * LENC) out[tid] = 0.0f;
    if (tid < TWID_ELEMS) {
        int e    = tid & 15;
        int lane = (tid >> 4) & 31;
        int ct   = (tid >> 9) & 15;   // output column tile
        int c    = (tid >> 13) & 7;   // K chunk (32 wide)
        int m = (c << 5) + ((lane >> 4) << 4) + e;   // DFT input index
        int k = (ct << 4) + (lane & 15);             // DFT output index
        int ph = (m * k) & (MC - 1);                 // (m*k) mod 256, exact
        float ang = 6.283185307179586f * (float)ph * (1.0f / (float)MC);
        float sv, cv;
        __sincosf(ang, &sv, &cv);
        cosT[tid] = (_Float16)cv;
        sinT[tid] = (_Float16)sv;
    }
}

// ---------------------------------------------------------------------------
// Kernel 1: batched 256-point complex DFT as WMMA GEMM.
// One workgroup (16 waves) = one 16-row tile x all 16 column tiles.
// The 16x256 complex X tile (32 KB) is staged into LDS exactly once via
// GLOBAL_LOAD_ASYNC_TO_LDS_B64 (ASYNCcnt) and reused by all 16 waves,
// cutting X HBM traffic 16x vs. per-wave gathering.
//   Re = xr.cos + xi.sin ; Im = xi.cos - xr.sin   (4 f32 WMMA accumulators)
// ---------------------------------------------------------------------------
__global__ __launch_bounds__(512) void insgt_dft_wmma(
        const float* __restrict__ X,
        const _Float16* __restrict__ cosT,
        const _Float16* __restrict__ sinT,
        float2* __restrict__ temp) {
    __shared__ float2 xtile[16 * MC];          // [row][m], 32 KB

    int tid     = threadIdx.x;
    int lane    = tid & 31;
    int tileCol = tid >> 5;                    // wave id = output column tile
    int tileRow = blockIdx.x;                  // 0..2047
    int r0      = tileRow << 4;
    int sb      = r0 >> 7;                     // s*8 + b (same for all 16 rows)
    int f0      = r0 & (F1C - 1);
    int b       = sb & 7;
    int s       = sb >> 3;

    // X[b, f, m, s, {re,im}] : float index = (((b*128+f)*256 + m)*32 + s)*2
    const float* xbase = X + ((size_t)((b * F1C + f0) * MC) * SC + (size_t)s) * 2;

    // --- cooperative async stage: 4096 x 8B transfers, 8 per thread ---
    for (int i = tid; i < 16 * MC; i += 512) {
        int row = i >> 8;
        int m   = i & (MC - 1);
        const void* g = (const void*)(xbase + (size_t)row * (MC * SC * 2)
                                            + (size_t)m * (SC * 2));
        unsigned la = lds_offset(&xtile[i]);
        asm volatile("global_load_async_to_lds_b64 %0, %1, off"
                     :: "v"(la), "v"(g) : "memory");
    }
    asm volatile("s_wait_asynccnt 0" ::: "memory");
    __syncthreads();

    // A-fragment row for this lane (lanes 0-15 and 16-31 both M = L%16)
    int halfsel = lane >> 4;
    const float2* xrow = &xtile[(lane & 15) << 8];

    v8f accPP = {}, accQQ = {}, accRR = {}, accUU = {};
    for (int c = 0; c < 8; ++c) {
        v16h a_re, a_im;
#pragma unroll
        for (int e = 0; e < 16; ++e) {
            // A k order used here (same as B tables): K = 16*(lane/16) + e
            int K = (halfsel << 4) + e;
            float2 xv = xrow[(c << 5) + K];
            a_re[e] = (_Float16)xv.x;
            a_im[e] = (_Float16)xv.y;
        }
        // B fragments: 32 contiguous bytes per lane (pre-swizzled tables, L2-resident)
        const v16h b_cos = *(const v16h*)(cosT + ((((c << 4) + tileCol) << 5 | lane) << 4));
        const v16h b_sin = *(const v16h*)(sinT + ((((c << 4) + tileCol) << 5 | lane) << 4));

        accPP = __builtin_amdgcn_wmma_f32_16x16x32_f16(false, a_re, false, b_cos,
                                                       (short)0, accPP, false, false);
        accQQ = __builtin_amdgcn_wmma_f32_16x16x32_f16(false, a_im, false, b_sin,
                                                       (short)0, accQQ, false, false);
        accRR = __builtin_amdgcn_wmma_f32_16x16x32_f16(false, a_im, false, b_cos,
                                                       (short)0, accRR, false, false);
        accUU = __builtin_amdgcn_wmma_f32_16x16x32_f16(false, a_re, false, b_sin,
                                                       (short)0, accUU, false, false);
    }

    // D layout: VGPR j -> M = j + 8*(lane/16), N = lane%16
    int kout = (tileCol << 4) + (lane & 15);
#pragma unroll
    for (int j = 0; j < 8; ++j) {
        int rr = r0 + j + (halfsel << 3);
        float re = accPP[j] + accQQ[j];
        float im = accRR[j] - accUU[j];
        temp[(size_t)rr * MC + kout] = make_float2(re, im);
    }
}

// ---------------------------------------------------------------------------
// Kernel 2: per-(s,b) fused ragged scatter-add -> 8192-pt inverse FFT -> OLA.
// Entire slice spectrum lives in LDS (64 KB of the 320 KB WGP pool).
// Scatter targets are bit-reversed so the 13-stage radix-2 DIT IFFT yields
// natural-order output with no permutation pass. ds_add_f32 handles the
// cross-bin index collisions; OLA uses global_atomic_add_f32.
// ---------------------------------------------------------------------------
__global__ __launch_bounds__(1024) void insgt_scatter_ifft_ola(
        const float2* __restrict__ temp,
        const float* __restrict__ gd,
        const int* __restrict__ li,
        const int* __restrict__ gi,
        float* __restrict__ out) {
    __shared__ float lre[NC];
    __shared__ float lim[NC];
    int tid = threadIdx.x;
    int sb  = blockIdx.x;             // s*8 + b
    int s   = sb >> 3;
    int b   = sb & 7;

    const float2* trow = temp + (size_t)sb * F1C * MC;
    // Pull the 256 KB spectrum block toward the caches ahead of the random
    // gather (global_prefetch_b8, one line per 16 float2).
    for (int i = tid * 16; i < F1C * MC; i += 1024 * 16)
        __builtin_prefetch(&trow[i], 0, 0);

    for (int i = tid; i < NC; i += 1024) { lre[i] = 0.0f; lim[i] = 0.0f; }
    __syncthreads();

    // Ragged gather + window + wrap-around scatter (bit-reversed target)
    for (int i = tid; i < F1C * LMAXC; i += 1024) {
        float w = gd[i];
        if (w != 0.0f) {
            int f  = i >> 10;
            int kk = li[i];
            int nn = gi[i];
            float2 cv = trow[(f << 8) + kk];
            int rv = (int)(__brev((unsigned)nn) >> 19);   // 13-bit reverse
            atomicAdd(&lre[rv], w * cv.x);
            atomicAdd(&lim[rv], w * cv.y);
        }
    }
    __syncthreads();

    // In-place radix-2 DIT inverse FFT (input bit-reversed, twiddle e^{+i*ang})
    for (int st = 0; st < 13; ++st) {
        int half = 1 << st;
        for (int p = tid; p < NC / 2; p += 1024) {
            int j    = p & (half - 1);
            int base = ((p >> st) << (st + 1)) + j;
            int i0 = base, i1 = base + half;
            float ang = 3.14159265358979f * (float)j / (float)half;
            float wi, wr;
            __sincosf(ang, &wi, &wr);
            float xr = lre[i1], xi = lim[i1];
            float tr = wr * xr - wi * xi;
            float ti = wr * xi + wi * xr;
            float ur = lre[i0], ui = lim[i0];
            lre[i1] = ur - tr;  lim[i1] = ui - ti;
            lre[i0] = ur + tr;  lim[i0] = ui + ti;
        }
        __syncthreads();
    }

    // Real part, 1/N scale, 50%-overlap-add, trim to LENGTH
    const float scale = 1.0f / (float)NC;
    int ola  = s * (NC / 2);
    float* orow = out + (size_t)b * LENC;
    for (int n = tid; n < NC; n += 1024) {
        int pos = ola + n;
        if (pos < LENC) atomicAdd(&orow[pos], lre[n] * scale);
    }
}

// ---------------------------------------------------------------------------
extern "C" void kernel_launch(void* const* d_in, const int* in_sizes, int n_in,
                              void* d_out, int out_size, void* d_ws, size_t ws_size,
                              hipStream_t stream) {
    const float* X  = (const float*)d_in[0];   // (1,4,2, 32768, 32, 2) f32
    const float* gd = (const float*)d_in[1];   // (128,1024) f32
    const int*   li = (const int*)d_in[2];     // (128,1024) i32
    const int*   gi = (const int*)d_in[3];     // (128,1024) i32
    // d_in[4]=slice_len, d_in[5]=length : compile-time constants here
    float* out = (float*)d_out;                // (1,4,2,131072) f32

    char* ws = (char*)d_ws;
    float2*   temp = (float2*)ws;                         // 67,108,864 B
    _Float16* cosT = (_Float16*)(ws + TEMP_BYTES);        // 131,072 B
    _Float16* sinT = cosT + TWID_ELEMS;                   // 131,072 B

    insgt_init<<<(BC * LENC + 255) / 256, 256, 0, stream>>>(out, cosT, sinT);
    insgt_dft_wmma<<<ROWS / 16, 512, 0, stream>>>(X, cosT, sinT, temp);
    insgt_scatter_ifft_ola<<<SC * BC, 1024, 0, stream>>>(temp, gd, li, gi, out);
}
